// MaskGenerator_11836929868231
// MI455X (gfx1250) — matrix operand order repaired and checked by reading, not compiled, not run on hardware
//
#include <hip/hip_runtime.h>
#include <hip/hip_bf16.h>
#include <math.h>

typedef __attribute__((ext_vector_type(16))) _Float16 v16h;
typedef __attribute__((ext_vector_type(8)))  float    v8f;
typedef __attribute__((ext_vector_type(2)))  float    v2f;
typedef __attribute__((ext_vector_type(4)))  unsigned u32x4;
typedef __attribute__((ext_vector_type(8)))  int      i32x8;
typedef __attribute__((ext_vector_type(4)))  int      i32x4;

#define B_    256
#define L_    512
#define F_    64
#define NOUT  32768   // L_*F_
#define KH    256     // hidden dim (2*H)
#define WTS   17      // padded LDS row stride (dwords) for weight tiles
#define WTSZ  (KH * WTS)   // dwords per staged weight matrix (256*17)

#if __has_builtin(__builtin_amdgcn_wmma_f32_16x16x4_f32)
#define HAVE_WMMA_F32X4 1
#endif

#if __has_builtin(__builtin_amdgcn_tensor_load_to_lds) && __has_builtin(__builtin_amdgcn_s_wait_tensorcnt)
#define USE_TDM 1
#endif

// ---------------- deterministic RNG helpers (hash-based) ----------------
__device__ __forceinline__ unsigned pcg(unsigned v) {
    v = v * 747796405u + 2891336453u;
    unsigned w = ((v >> ((v >> 28) + 4u)) ^ v) * 277803737u;
    return (w >> 22) ^ w;
}
__device__ __forceinline__ float u01(unsigned h) {
    return (float)(h >> 8) * (1.0f / 16777216.0f);
}
__device__ __forceinline__ float nrmf(unsigned i) {
    float u1 = u01(pcg(i * 2u + 0x9E3779B9u));
    float u2 = u01(pcg(i * 2u + 0x85EBCA6Bu));
    u1 = fmaxf(u1, 1e-7f);
    return sqrtf(-2.0f * __logf(u1)) * __cosf(6.28318530718f * u2);
}
__device__ __forceinline__ float d_sigmoid(float v) { return 1.0f / (1.0f + __expf(-v)); }
__device__ __forceinline__ float d_softplus(float v) { return (v > 20.0f) ? v : log1pf(__expf(v)); }

// ---------------- init: scalars + warp params ----------------
// S[0]=s_mask S[1]=s_noise S[2]=s_shift S[3]=s_scale S[4..6]=wa,wb,wc S[7]=ratio S[8]=s_mix
__global__ void init_kernel(const float* pm, const float* pn, const float* psh, const float* psc,
                            float* S, int* WP, int* SHv) {
    int t = threadIdx.x;
    float s_shift = d_sigmoid(psh[0]);
    int wsteps = (int)((float)L_ * 0.1f * s_shift);
    if (t == 0) {
        float s_mask  = d_sigmoid(pm[0]);
        float s_noise = d_sigmoid(pn[0]);
        float s_scale = d_sigmoid(psc[0]);
        float s_mix   = 1.0f - s_shift;
        float wa = fminf(fmaxf(1.0f - s_mix - s_shift, 0.1f), 0.8f);
        float wb = s_mix * 0.5f;
        float wc = s_shift * 0.5f;
        float tot = wa + wb + wc;
        S[0] = s_mask; S[1] = s_noise; S[2] = s_shift; S[3] = s_scale;
        S[4] = wa / tot; S[5] = wb / tot; S[6] = wc / tot;
        S[7] = fminf(s_mix * 0.1f, 0.5f); S[8] = s_mix;
    }
    if (t < B_) {
        int range = L_ - 2 * wsteps; if (range < 1) range = 1;
        WP[t]  = wsteps + (int)(pcg(0x1111A000u + t) % (unsigned)range);
        SHv[t] = -wsteps + (int)(pcg(0x2222B000u + t) % (unsigned)(2 * wsteps + 1));
    }
}

// ---------------- z ~ N(0,1), (256,64) ----------------
__global__ void genz_kernel(float* Z) {
    int i = blockIdx.x * 256 + threadIdx.x;          // 64 blocks * 256 = 16384
    Z[i] = nrmf(0x33330000u + (unsigned)i);
}

// ---------------- generic f32 GEMM tile kernel (1 wave/block, 16x16 tile) ----------------
// out(256 x N) = act(A(256 x K) @ W(K x N) + bias)
// act==0: leaky relu ; act==2: 1 + (softplus(v)-0.5)*0.2*s_scale
__global__ void gemm_act_kernel(const float* __restrict__ A, const float* __restrict__ W,
                                const float* __restrict__ bias, float* __restrict__ out,
                                int K, int N, int act, const float* __restrict__ S) {
    const int lane = threadIdx.x & 31;
    const int tm = blockIdx.x & 15;     // 16 row tiles (256 rows)
    const int tn = blockIdx.x >> 4;
    const int m  = lane & 15;
    const int hi = lane >> 4;
    const int ra = tm * 16 + m;         // A row for this lane
    const int cn = tn * 16 + m;         // W col for this lane
    v8f c = {};
#ifdef HAVE_WMMA_F32X4
    for (int kk = 0; kk < K; kk += 4) {
        v2f av, bv;
        av.x = A[(size_t)ra * K + kk + 2 * hi + 0];
        av.y = A[(size_t)ra * K + kk + 2 * hi + 1];
        bv.x = W[(size_t)(kk + 2 * hi + 0) * N + cn];
        bv.y = W[(size_t)(kk + 2 * hi + 1) * N + cn];
        c = __builtin_amdgcn_wmma_f32_16x16x4_f32(false, av, false, bv, (short)0, c,
                                                  false, false);
    }
#else
    #pragma unroll
    for (int r = 0; r < 8; ++r) {
        int row = tm * 16 + hi * 8 + r;
        float acc = 0.0f;
        for (int k = 0; k < K; ++k) acc += A[(size_t)row * K + k] * W[(size_t)k * N + cn];
        c[r] = acc;
    }
#endif
    const float s_scale = S[3];
    const float bb = bias[cn];
    #pragma unroll
    for (int r = 0; r < 8; ++r) {
        int row = tm * 16 + hi * 8 + r;
        float v = c[r] + bb;
        if (act == 0) v = (v > 0.0f) ? v : 0.2f * v;
        else          v = 1.0f + (d_softplus(v) - 0.5f) * 0.2f * s_scale;
        out[(size_t)row * N + cn] = v;
    }
}

// ---------------- convert h (256x256 f32) -> f16 ----------------
__global__ void cvt_kernel(const float* __restrict__ H3, _Float16* __restrict__ H16) {
    for (int i = blockIdx.x * 256 + threadIdx.x; i < 256 * 256; i += 64 * 256)
        H16[i] = (_Float16)H3[i];
}

// ---------------- freq augmentation: 512-pt FFT -> mask -> IFFT, 8 signals/block ----------------
#define FSIG 8
__global__ void __launch_bounds__(256) freq_kernel(const float* __restrict__ x,
                                                   float* __restrict__ out,
                                                   const float* __restrict__ S) {
    __shared__ float2 sh[L_ * FSIG];                 // 32 KB LDS
    const int tid = threadIdx.x;
    const int b   = blockIdx.x >> 3;                 // 2048 blocks: 256 b * 8 fgroups
    const int f0  = (blockIdx.x & 7) * FSIG;
    const float ratio = S[7];
    const float s_mix = S[8];
    // load with bit-reversal on l
    for (int i = tid; i < L_ * FSIG; i += 256) {
        int l = i >> 3, j = i & 7;
        float v = x[((size_t)b * L_ + l) * F_ + f0 + j];
        int r = (int)(__brev((unsigned)l) >> 23);
        sh[r * FSIG + j] = make_float2(v, 0.0f);
    }
    // forward FFT (DIT, bit-reversed input -> natural output)
    for (int st = 0; st < 9; ++st) {
        int half = 1 << st;
        __syncthreads();
        for (int i = tid; i < (L_ / 2) * FSIG; i += 256) {
            int bf = i >> 3, j = i & 7;
            int grp = bf / half, pos = bf - grp * half;
            int i0 = (grp * 2 * half + pos) * FSIG + j;
            int i1 = i0 + half * FSIG;
            float ang = -3.14159265358979f * (float)pos / (float)half;
            float cc = __cosf(ang), ss = __sinf(ang);
            float2 u = sh[i0], v = sh[i1];
            float vr = v.x * cc - v.y * ss;
            float vi = v.x * ss + v.y * cc;
            sh[i0] = make_float2(u.x + vr, u.y + vi);
            sh[i1] = make_float2(u.x - vr, u.y - vi);
        }
    }
    // frequency-domain random keep mask
    __syncthreads();
    for (int i = tid; i < L_ * FSIG; i += 256) {
        int l = i >> 3, j = i & 7;
        unsigned h = pcg(0x51F7C000u ^ (unsigned)((b * L_ + l) * F_ + f0 + j));
        float keep = (u01(h) > ratio) ? 1.0f : 0.0f;
        float2 v = sh[l * FSIG + j];
        sh[l * FSIG + j] = make_float2(v.x * keep, v.y * keep);
    }
    // bit-reverse permute again for the inverse pass
    __syncthreads();
    for (int i = tid; i < L_ * FSIG; i += 256) {
        int l = i >> 3, j = i & 7;
        int r = (int)(__brev((unsigned)l) >> 23);
        if (r > l) {
            float2 a = sh[l * FSIG + j];
            float2 c = sh[r * FSIG + j];
            sh[l * FSIG + j] = c;
            sh[r * FSIG + j] = a;
        }
    }
    // inverse FFT (positive twiddles)
    for (int st = 0; st < 9; ++st) {
        int half = 1 << st;
        __syncthreads();
        for (int i = tid; i < (L_ / 2) * FSIG; i += 256) {
            int bf = i >> 3, j = i & 7;
            int grp = bf / half, pos = bf - grp * half;
            int i0 = (grp * 2 * half + pos) * FSIG + j;
            int i1 = i0 + half * FSIG;
            float ang = 3.14159265358979f * (float)pos / (float)half;
            float cc = __cosf(ang), ss = __sinf(ang);
            float2 u = sh[i0], v = sh[i1];
            float vr = v.x * cc - v.y * ss;
            float vi = v.x * ss + v.y * cc;
            sh[i0] = make_float2(u.x + vr, u.y + vi);
            sh[i1] = make_float2(u.x - vr, u.y - vi);
        }
    }
    __syncthreads();
    for (int i = tid; i < L_ * FSIG; i += 256) {
        int l = i >> 3, j = i & 7;
        float res = sh[l * FSIG + j].x * (1.0f / 512.0f);
        size_t o = ((size_t)b * L_ + l) * F_ + f0 + j;
        out[o] = (s_mix < 0.01f) ? x[o] : res;
    }
}

#ifdef USE_TDM
// Issue one TDM DMA: 2D tile (16 cols x 256 k-rows of f32) from a (256 x 32768)
// row-major tensor into LDS, with 1-dword padding after each 64B row so the staged
// layout has a 17-dword stride (bank-conflict-free for the 16-lane read pattern).
__device__ __forceinline__ void tdm_load_tile(const float* gsrc, unsigned lds_byte_off) {
    unsigned long long ga = (unsigned long long)(size_t)gsrc;
    u32x4 g0;
    g0[0] = 1u;                                        // count=1, user descriptor
    g0[1] = lds_byte_off;                              // lds_addr
    g0[2] = (unsigned)(ga & 0xFFFFFFFFu);              // global_addr[31:0]
    g0[3] = (unsigned)((ga >> 32) & 0x01FFFFFFu)       // global_addr[56:32]
          | (2u << 30);                                // type=2 (image)
    i32x8 g1;
    g1[0] = (int)((2u << 16)      // data_size = 4 bytes
                | (1u << 20)      // pad_enable
                | (3u << 22));    // pad_interval: every 64 bytes (16 dwords)
                                  // pad_amount field 0 => 1 dword
    g1[1] = (int)0x80000000u;     // tensor_dim0 = 32768 (low 16 bits at [63:48])
    g1[2] = (int)(256u << 16);    // tensor_dim0 hi=0, tensor_dim1 = 256
    g1[3] = (int)(16u << 16);     // tensor_dim1 hi=0, tile_dim0 = 16
    g1[4] = (int)256;             // tile_dim1 = 256, tile_dim2 = 0
    g1[5] = (int)32768;           // tensor_dim0_stride low 32
    g1[6] = 0;                    // stride0 hi, stride1 low16 (stride1=0x800000)
    g1[7] = (int)0x80;            // stride1 bits [47:16]
    i32x4 z = {0, 0, 0, 0};
#if __clang_major__ >= 23
    i32x8 z8 = {};
    __builtin_amdgcn_tensor_load_to_lds(g0, g1, z, z, z8, 0);
#else
    __builtin_amdgcn_tensor_load_to_lds(g0, g1, z, z, 0);
#endif
}
#endif

// ---------------- fused dual-GEMM (mask & noise) + full combine epilogue ----------------
// Each wave owns one 16x16 tile of the (256 x 32768) outputs; K=256 in 8 steps of
// v_wmma_f32_16x16x32_f16, two accumulators sharing A loads. The block's weight
// tiles (shared by all 8 waves) are staged into LDS once by the Tensor Data Mover
// (TENSORcnt + s_wait_tensorcnt), then read via ds_load + cvt to f16. Epilogue
// applies sigmoid/softplus, noise RNG, per-(b,f) scale, time-warp gather, and
// blends with the freq result previously staged in `out` (same-thread r/w).
__global__ void __launch_bounds__(256) fused_gemm_combine(
    const _Float16* __restrict__ A16, const float* __restrict__ wm, const float* __restrict__ bm,
    const float* __restrict__ wn, const float* __restrict__ bn, const float* __restrict__ x,
    const float* __restrict__ SC, const int* __restrict__ WP, const int* __restrict__ SHv,
    const float* __restrict__ S, float* __restrict__ out) {
    const int lane = threadIdx.x & 31;
    const int wave = threadIdx.x >> 5;
    const int tn = blockIdx.x >> 1;                       // 2048 column tiles
    const int tm = ((blockIdx.x & 1) << 3) + wave;        // 16 row tiles
    const int hi = lane >> 4;
    const int n  = lane & 15;
    const int col = tn * 16 + n;
    const int rowA = tm * 16 + n;
    v8f cm = {}; v8f cn = {};

#ifdef USE_TDM
    __shared__ float WT[2 * WTSZ];                        // 2 x (256x17) dwords = 34 KB
    if (threadIdx.x < 32) {                               // wave 0 drives the TDM
        unsigned lds0 = (unsigned)(unsigned long long)
            (__attribute__((address_space(3))) float*)&WT[0];
        tdm_load_tile(wm + (size_t)tn * 16, lds0);
        tdm_load_tile(wn + (size_t)tn * 16, lds0 + WTSZ * 4u);
        __builtin_amdgcn_s_wait_tensorcnt(0);
    }
    __syncthreads();
    #pragma unroll
    for (int k0 = 0; k0 < KH; k0 += 32) {
        v16h a;
        const _Float16* Ap = A16 + (size_t)rowA * KH + k0 + hi * 16;
        #pragma unroll
        for (int e = 0; e < 8; ++e) { a[e] = Ap[e]; a[8 + e] = Ap[8 + e]; }
        const int kb = k0 + hi * 16;
        v16h bmh, bnh;
        #pragma unroll
        for (int e = 0; e < 16; ++e) {
            bmh[e] = (_Float16)WT[(kb + e) * WTS + n];
            bnh[e] = (_Float16)WT[WTSZ + (kb + e) * WTS + n];
        }
        cm = __builtin_amdgcn_wmma_f32_16x16x32_f16(false, a, false, bmh, (short)0, cm,
                                                    false, false);
        cn = __builtin_amdgcn_wmma_f32_16x16x32_f16(false, a, false, bnh, (short)0, cn,
                                                    false, false);
    }
#else
    #pragma unroll
    for (int k0 = 0; k0 < KH; k0 += 32) {
        v16h a;
        const _Float16* Ap = A16 + (size_t)rowA * KH + k0 + hi * 16;
        #pragma unroll
        for (int e = 0; e < 8; ++e) { a[e] = Ap[e]; a[8 + e] = Ap[8 + e]; }
        const int kb = k0 + hi * 16;
        const float* wmp = wm + (size_t)kb * NOUT + col;
        const float* wnp = wn + (size_t)kb * NOUT + col;
        v16h bmh, bnh;
        #pragma unroll
        for (int e = 0; e < 16; ++e) {
            bmh[e] = (_Float16)wmp[(size_t)e * NOUT];
            bnh[e] = (_Float16)wnp[(size_t)e * NOUT];
        }
        if (k0 + 32 < KH) {
            __builtin_prefetch(wmp + (size_t)32 * NOUT, 0, 1);
            __builtin_prefetch(wnp + (size_t)32 * NOUT, 0, 1);
        }
        cm = __builtin_amdgcn_wmma_f32_16x16x32_f16(false, a, false, bmh, (short)0, cm,
                                                    false, false);
        cn = __builtin_amdgcn_wmma_f32_16x16x32_f16(false, a, false, bnh, (short)0, cn,
                                                    false, false);
    }
#endif

    const float s_mask = S[0], s_noise = S[1];
    const float wa = S[4], wb = S[5], wc = S[6];
    const float bmb = bm[col], bnb = bn[col];
    const int l = col >> 6, f = col & 63;
    #pragma unroll
    for (int r = 0; r < 8; ++r) {
        const int row = tm * 16 + hi * 8 + r;             // batch index
        float gm = cm[r] + bmb;
        float gn = cn[r] + bnb;
        float mask = 1.0f - (1.0f - d_sigmoid(gm)) * s_mask * 0.3f;
        float nz = nrmf(0x77770000u + (unsigned)((row * L_ + l) * F_ + f));
        float noise = nz * d_softplus(gn) * s_noise * 0.05f;
        float sc = SC[row * F_ + f];
        size_t xo = ((size_t)row * L_ + l) * F_ + f;
        float xv = x[xo];
        int wp = WP[row], sv = SHv[row];
        int idx;
        if (sv > 0)      idx = (l >= wp) ? min(l + sv, L_ - 1) : l;
        else if (sv < 0) idx = (l >= wp + sv && l < L_ + sv) ? (l - sv) : l;
        else             idx = l;
        float wv = x[((size_t)row * L_ + idx) * F_ + f];
        float fv = out[xo];                                // freq result staged earlier
        out[xo] = (xv * mask * sc + noise) * wa + fv * wb + wv * wc;
    }
}

// ---------------- host launcher ----------------
extern "C" void kernel_launch(void* const* d_in, const int* in_sizes, int n_in,
                              void* d_out, int out_size, void* d_ws, size_t ws_size,
                              hipStream_t stream) {
    const float* x   = (const float*)d_in[0];
    const float* w1  = (const float*)d_in[2];
    const float* b1  = (const float*)d_in[3];
    const float* w2  = (const float*)d_in[4];
    const float* b2  = (const float*)d_in[5];
    const float* w3  = (const float*)d_in[6];
    const float* b3  = (const float*)d_in[7];
    const float* wm  = (const float*)d_in[8];
    const float* bm  = (const float*)d_in[9];
    const float* wn  = (const float*)d_in[10];
    const float* bn  = (const float*)d_in[11];
    const float* wsw = (const float*)d_in[12];
    const float* bs  = (const float*)d_in[13];
    const float* pm  = (const float*)d_in[14];
    const float* pn  = (const float*)d_in[15];
    const float* psh = (const float*)d_in[16];
    const float* psc = (const float*)d_in[17];
    float* out = (float*)d_out;

    // workspace layout (< 1 MB)
    float* S   = (float*)d_ws;                 // 16 scalars
    int*   WP  = (int*)(S + 16);               // 256
    int*   SHv = WP + 256;                     // 256
    float* Z   = (float*)(SHv + 256);          // 256*64
    float* H1  = Z  + 256 * 64;                // 256*128
    float* H2  = H1 + 256 * 128;               // 256*256
    float* H3  = H2 + 256 * 256;               // 256*256
    float* SC  = H3 + 256 * 256;               // 256*64
    _Float16* H16 = (_Float16*)(SC + 256 * 64);// 256*256 halves

    init_kernel<<<1, 256, 0, stream>>>(pm, pn, psh, psc, S, WP, SHv);
    genz_kernel<<<64, 256, 0, stream>>>(Z);
    // MLP: 64->128->256->256 (leaky), then scale head 256->64
    gemm_act_kernel<<<16 * 8,  32, 0, stream>>>(Z,  w1,  b1, H1,  64, 128, 0, S);
    gemm_act_kernel<<<16 * 16, 32, 0, stream>>>(H1, w2,  b2, H2, 128, 256, 0, S);
    gemm_act_kernel<<<16 * 16, 32, 0, stream>>>(H2, w3,  b3, H3, 256, 256, 0, S);
    gemm_act_kernel<<<16 * 4,  32, 0, stream>>>(H3, wsw, bs, SC, 256,  64, 2, S);
    cvt_kernel<<<64, 256, 0, stream>>>(H3, H16);
    // stage freq-augmented signal into d_out
    freq_kernel<<<B_ * 8, 256, 0, stream>>>(x, out, S);
    // fused dual-GEMM (TDM-staged weights) + combine, overwrites d_out in place
    fused_gemm_combine<<<4096, 256, 0, stream>>>(H16, wm, bm, wn, bn, x, SC, WP, SHv, S, out);
}
